// DDSP_Decoder_17635135717893
// MI455X (gfx1250) — compile-verified, matrix operand order checked by reading, not run on hardware
//
#include <hip/hip_runtime.h>
#include <hip/hip_bf16.h>
#include <math.h>

// ---------------------------------------------------------------------------
// DDSP decoder for MI455X (gfx1250, wave32, WMMA).
// Dense layers -> v_wmma_f32_16x16x32_f16 tiled GEMM (f16 LDS tiles, f32 acc).
// GRU: GI precomputed by WMMA GEMM; serial part persistent-workgroup w/ LDS h.
// Reverb: time-domain causal conv, LDS staging via async global->LDS loads.
// ---------------------------------------------------------------------------

typedef _Float16 half16 __attribute__((ext_vector_type(16)));
typedef _Float16 half4v __attribute__((ext_vector_type(4)));
typedef float    float8 __attribute__((ext_vector_type(8)));

#define PI_F 3.14159265358979f

// ----------------------------- async->LDS helpers ---------------------------
#if __has_builtin(__builtin_amdgcn_global_load_async_to_lds_b128)
#define DDSP_HAS_ASYNC 1
typedef int ddsp_v4i __attribute__((vector_size(4 * sizeof(int))));
typedef __attribute__((address_space(3))) ddsp_v4i ddsp_v4i_lds;
#endif

__device__ __forceinline__ void ddsp_copy16(const float* g, float* l) {
#ifdef DDSP_HAS_ASYNC
  __builtin_amdgcn_global_load_async_to_lds_b128((ddsp_v4i*)g, (ddsp_v4i_lds*)l, 0, 0);
#else
  float4 v = *(const float4*)g;
  *(float4*)l = v;
#endif
}

__device__ __forceinline__ void ddsp_async_wait() {
#ifdef DDSP_HAS_ASYNC
#if __has_builtin(__builtin_amdgcn_s_wait_asynccnt)
  __builtin_amdgcn_s_wait_asynccnt(0);
#else
  asm volatile("s_wait_asynccnt 0" ::: "memory");
#endif
#endif
}

// ----------------------------- WMMA GEMM -----------------------------------
// C[M x N] = A[M x Ka(f32)] * W[N x Kw(f32)]^T + bias + c_add
// Requirements: Ka % 32 == 0, Kw % 4 == 0, lda >= Ka; if Kw < Ka then A's
// columns [Kw..Ka) are zero and W rows are Kw-padded (host pre-pads).
// Block: 128 threads (4 waves), 64x64 tile, each wave 32x32 (4 WMMAs / kstep).
// OOB rows/cols are index-CLAMPED (they only feed OOB outputs, guarded at
// store), so all tile staging is branchless vectorized global_load_b128.

#define GEMM_LDH 40   // f16 LDS row stride (avoids bank conflicts)

__device__ __forceinline__ half16 ddsp_frag(const _Float16* base, int row0) {
  const int lane = threadIdx.x & 31;
  const int r    = row0 + (lane & 15);
  const int kb   = (lane & 16) ? 8 : 0;     // documented CDNA5 16-bit A/B layout
  const _Float16* p = base + r * GEMM_LDH;
  half16 f;
#pragma unroll
  for (int i = 0; i < 8; ++i) { f[i] = p[kb + i]; f[8 + i] = p[16 + kb + i]; }
  return f;
}

__global__ __launch_bounds__(128)
void k_gemm(const float* __restrict__ A, int lda, int Ka,
            const float* __restrict__ W, int Kw,
            const float* __restrict__ bias,
            float* __restrict__ C, int ldc, int col0,
            int M, int N, float c_add) {
  __shared__ _Float16 sA[64 * GEMM_LDH];
  __shared__ _Float16 sB[64 * GEMM_LDH];

  const int rowBase = blockIdx.x * 64;
  const int colBase = blockIdx.y * 64;
  const int wave = threadIdx.x >> 5;
  const int wr = wave >> 1, wc = wave & 1;        // 2x2 wave grid, 32x32 each

  const int idx0 = threadIdx.x;                   // staging base index
  float8 acc00 = {0,0,0,0,0,0,0,0};
  float8 acc01 = {0,0,0,0,0,0,0,0};
  float8 acc10 = {0,0,0,0,0,0,0,0};
  float8 acc11 = {0,0,0,0,0,0,0,0};

  for (int kt = 0; kt < Ka; kt += 32) {
    float4 va[4], vb[4];
#pragma unroll
    for (int i = 0; i < 4; ++i) {                 // branchless staging loads
      int idx = idx0 + i * 128;
      int r = idx >> 3, c4 = idx & 7;
      int gk = kt + c4 * 4;
      int gr = rowBase + r; gr = (gr < M) ? gr : (M - 1);        // clamp
      va[i] = *(const float4*)(A + (size_t)gr * lda + gk);
      int gn = colBase + r; gn = (gn < N) ? gn : (N - 1);        // clamp
      vb[i] = *(const float4*)(W + (size_t)gn * Kw + gk);
    }
    __syncthreads();                              // prev-iter frag reads done
#pragma unroll
    for (int i = 0; i < 4; ++i) {                 // f32->f16, packed b64 stores
      int idx = idx0 + i * 128;
      int r = idx >> 3, c4 = idx & 7;
      half4v ha = { (_Float16)va[i].x, (_Float16)va[i].y,
                    (_Float16)va[i].z, (_Float16)va[i].w };
      *(half4v*)(&sA[r * GEMM_LDH + c4 * 4]) = ha;
      half4v hb = { (_Float16)vb[i].x, (_Float16)vb[i].y,
                    (_Float16)vb[i].z, (_Float16)vb[i].w };
      *(half4v*)(&sB[r * GEMM_LDH + c4 * 4]) = hb;
    }
    __syncthreads();

    half16 a0 = ddsp_frag(sA, wr * 32);
    half16 a1 = ddsp_frag(sA, wr * 32 + 16);
    half16 b0 = ddsp_frag(sB, wc * 32);
    half16 b1 = ddsp_frag(sB, wc * 32 + 16);

    acc00 = __builtin_amdgcn_wmma_f32_16x16x32_f16(false, a0, false, b0, (short)0, acc00, false, false);
    acc01 = __builtin_amdgcn_wmma_f32_16x16x32_f16(false, a0, false, b1, (short)0, acc01, false, false);
    acc10 = __builtin_amdgcn_wmma_f32_16x16x32_f16(false, a1, false, b0, (short)0, acc10, false, false);
    acc11 = __builtin_amdgcn_wmma_f32_16x16x32_f16(false, a1, false, b1, (short)0, acc11, false, false);
  }

  const int lane = threadIdx.x & 31;
  const int colL = lane & 15;
  const int rb   = (lane & 16) ? 8 : 0;
  const bool fullTile = (rowBase + 64 <= M) && (colBase + 64 <= N);  // uniform
#pragma unroll
  for (int ti = 0; ti < 2; ++ti) {
#pragma unroll
    for (int tj = 0; tj < 2; ++tj) {
      float8 acc = (ti == 0) ? (tj == 0 ? acc00 : acc01)
                             : (tj == 0 ? acc10 : acc11);
      int r0 = rowBase + wr * 32 + ti * 16;
      int c0 = colBase + wc * 32 + tj * 16;
      int gc = c0 + colL;
      if (fullTile) {                             // common case: no guards
        float bb = bias[gc] + c_add;
        float* cp = C + (size_t)(r0 + rb) * ldc + col0 + gc;
#pragma unroll
        for (int v = 0; v < 8; ++v) cp[(size_t)v * ldc] = acc[v] + bb;
      } else {                                    // edge tiles
#pragma unroll
        for (int v = 0; v < 8; ++v) {
          int gr = r0 + rb + v;
          if (gr < M && gc < N)
            C[(size_t)gr * ldc + col0 + gc] = acc[v] + bias[gc] + c_add;
        }
      }
    }
  }
}

// --------------------------- small dense kernels ----------------------------
__global__ void k_affine1(const float* __restrict__ x, const float* __restrict__ W,
                          const float* __restrict__ b, float* __restrict__ Y) {
  int t = blockIdx.x * 256 + threadIdx.x;
  if (t >= 750 * 512) return;
  int r = t >> 9, n = t & 511;
  Y[t] = x[r] * W[n] + b[n];
}

__global__ void k_gemm16(const float* __restrict__ X, const float* __restrict__ W,
                         const float* __restrict__ b, float* __restrict__ Y) {
  int t = blockIdx.x * 256 + threadIdx.x;
  if (t >= 750 * 512) return;
  int r = t >> 9, n = t & 511;
  const float* x = X + r * 16;
  const float* w = W + n * 16;
  float acc = b[n];
#pragma unroll
  for (int k = 0; k < 16; ++k) acc += x[k] * w[k];
  Y[t] = acc;
}

// zero-pad out_mlp.W1 (512 x 514) -> Wpad (512 x 544)
__global__ void k_padW(const float* __restrict__ W, float* __restrict__ Wp) {
  int t = blockIdx.x * 256 + threadIdx.x;
  if (t >= 512 * 544) return;
  int n = t / 544, c = t - n * 544;
  Wp[t] = (c < 514) ? W[n * 514 + c] : 0.f;
}

// LayerNorm(512) + leaky_relu(0.01); X: row stride ldx, Y: row stride ldy + col0
__global__ __launch_bounds__(256)
void k_ln(const float* __restrict__ X, int ldx,
          const float* __restrict__ g, const float* __restrict__ be,
          float* __restrict__ Y, int ldy, int col0) {
  __shared__ float rs[256], rq[256];
  int row = blockIdx.x, tid = threadIdx.x;
  const float* x = X + (size_t)row * ldx;
  float v0 = x[tid], v1 = x[tid + 256];
  rs[tid] = v0 + v1; rq[tid] = v0 * v0 + v1 * v1;
  __syncthreads();
  for (int s = 128; s > 0; s >>= 1) {
    if (tid < s) { rs[tid] += rs[tid + s]; rq[tid] += rq[tid + s]; }
    __syncthreads();
  }
  float m    = rs[0] * (1.f / 512.f);
  float var  = rq[0] * (1.f / 512.f) - m * m;
  float rinv = rsqrtf(var + 1e-5f);
  float* y = Y + (size_t)row * ldy + col0;
#pragma unroll
  for (int j = 0; j < 2; ++j) {
    int i = tid + j * 256;
    float v = (x[i] - m) * rinv * g[i] + be[i];
    y[i] = (v > 0.f) ? v : 0.01f * v;
  }
}

__global__ void k_transp(const float* __restrict__ W, float* __restrict__ WT) {
  int t = blockIdx.x * 256 + threadIdx.x;
  if (t >= 1536 * 512) return;
  int n = t / 512, k = t - n * 512;
  WT[k * 1536 + n] = W[t];
}

// ------------------------------- GRU (serial) -------------------------------
// grid = 2 (batch), block = 768. GI: 750x1536 (b_ih included).
// H2: 750x544; cols 512/513 = pitch/loud, 514..543 zero (K-pad for out_mlp).
__global__ __launch_bounds__(768)
void k_gru(const float* __restrict__ GI, const float* __restrict__ WhhT,
           const float* __restrict__ b_hh, const float* __restrict__ pitch,
           const float* __restrict__ loud, float* __restrict__ H2) {
  __shared__ float h[512];
  __shared__ float gh[1536];
  int b = blockIdx.x;
  for (int i = threadIdx.x; i < 512; i += blockDim.x) h[i] = 0.f;
  __syncthreads();
  for (int f = 0; f < 375; ++f) {
    int row = b * 375 + f;
    for (int n = threadIdx.x; n < 1536; n += blockDim.x) {
      float acc = b_hh[n];
      const float* wt = WhhT + n;
      for (int k = 0; k < 512; ++k) acc += wt[k * 1536] * h[k];
      gh[n] = acc;
    }
    __syncthreads();
    const float* gi = GI + (size_t)row * 1536;
    for (int n = threadIdx.x; n < 512; n += blockDim.x) {
      float r  = 1.f / (1.f + __expf(-(gi[n] + gh[n])));
      float z  = 1.f / (1.f + __expf(-(gi[512 + n] + gh[512 + n])));
      float nn = tanhf(gi[1024 + n] + r * gh[1024 + n]);
      float hn = (1.f - z) * nn + z * h[n];
      h[n] = hn;
      H2[(size_t)row * 544 + n] = hn;
    }
    for (int c = 512 + (int)threadIdx.x; c < 544; c += blockDim.x) {
      float v = 0.f;
      if (c == 512) v = pitch[row];
      else if (c == 513) v = loud[row];
      H2[(size_t)row * 544 + c] = v;
    }
    __syncthreads();
  }
}

// ------------------------- projections / scaling ----------------------------
__device__ __forceinline__ float ddsp_scale(float x) {
  float s = 1.f / (1.f + __expf(-x));
  return 2.f * __powf(s, 2.30258509f) + 1e-7f;   // 2*sigmoid^ln(10)+1e-7
}

// p0: 750x101 -> total_amp (d_out) + normalized amps (750x100)
__global__ __launch_bounds__(128)
void k_proj0(const float* __restrict__ P0, const float* __restrict__ pitch,
             float* __restrict__ amps, float* __restrict__ total_out) {
  __shared__ float sa[128];
  __shared__ float s_tot, s_sum;
  int row = blockIdx.x, tid = threadIdx.x;
  float p = pitch[row];
  float a = 0.f;
  if (tid < 100) {
    float s  = ddsp_scale(P0[row * 101 + 1 + tid]);
    float aa = ((p * (float)(tid + 1)) < 24000.f ? 1.f : 0.f) + 1e-4f;
    a = s * aa;
  }
  sa[tid] = a;
  if (tid == 0) {
    s_tot = ddsp_scale(P0[row * 101]);
    total_out[row] = s_tot;
  }
  __syncthreads();
  for (int s = 64; s > 0; s >>= 1) {
    if (tid < s) sa[tid] += sa[tid + s];
    __syncthreads();
  }
  if (tid == 0) s_sum = sa[0];
  __syncthreads();
  if (tid < 100) amps[row * 100 + tid] = a / s_sum * s_tot;
}

__global__ void k_proj1(const float* __restrict__ P1raw, float* __restrict__ P1s) {
  int t = blockIdx.x * 256 + threadIdx.x;
  if (t >= 750 * 65) return;
  P1s[t] = ddsp_scale(P1raw[t]);                 // -5 already folded in GEMM
}

// --------------------------- harmonic oscillator ----------------------------
__global__ void k_base(const float* __restrict__ pitch, float* __restrict__ base) {
  if (threadIdx.x < 2) {
    int b = threadIdx.x;
    float acc = 0.f;
    for (int f = 0; f < 375; ++f) {
      base[b * 375 + f] = acc;
      acc += 512.f * 2.f * PI_F * pitch[b * 375 + f] / 48000.f;
    }
  }
}

__global__ __launch_bounds__(256)
void k_harm(const float* __restrict__ pitch, const float* __restrict__ base,
            const float* __restrict__ amps, float* __restrict__ out) {
  __shared__ float aC[100], aN[100];
  int row = blockIdx.x;              // b*375+f
  int f   = row % 375;
  int nrow = (f + 1 < 375) ? row + 1 : row;
  for (int i = threadIdx.x; i < 100; i += 256) {
    aC[i] = amps[row * 100 + i];
    aN[i] = amps[nrow * 100 + i];
  }
  __syncthreads();
  float w  = 2.f * PI_F * pitch[row] / 48000.f;
  float ob = base[row];
  for (int i = threadIdx.x; i < 512; i += 256) {
    float omega = ob + (float)(i + 1) * w;
    float h1 = 0.5f * (1.f - __cosf((2.f * PI_F / 1024.f) * (float)i));
    float h2 = 0.5f * (1.f - __cosf((2.f * PI_F / 1024.f) * (float)(i + 512)));
    float acc = 0.f;
    for (int k = 0; k < 100; ++k)
      acc += __sinf(omega * (float)(k + 1)) * (aN[k] * h1 + aC[k] * h2);
    out[384000 + row * 512 + i] = acc;           // harmonic region of d_out
  }
}

// ------------------------------ noise branch --------------------------------
__global__ __launch_bounds__(128)
void k_noise(const float* __restrict__ noise_u, const float* __restrict__ P1,
             float* __restrict__ out) {
  __shared__ float ir0[128], kk[512], sg[512];
  int row = blockIdx.x, tid = threadIdx.x;
  const float* A = P1 + row * 65;
  {   // 128-pt inverse rfft of real half-spectrum (65 bins)
    int q = tid;
    float acc = A[0];
    for (int k = 1; k < 64; ++k)
      acc += 2.f * A[k] * __cosf((2.f * PI_F / 128.f) * (float)(k * q));
    acc += A[64] * __cosf(PI_F * (float)q);
    ir0[q] = acc * (1.f / 128.f);
  }
  __syncthreads();
#pragma unroll
  for (int j = 0; j < 4; ++j) {                  // windowed, rolled, padded IR
    int m = tid + 128 * j;
    int n = (m + 64) & 511;
    float v = 0.f;
    if (n < 128) {
      float win = 0.5f * (1.f - __cosf((2.f * PI_F / 128.f) * (float)n));
      v = ir0[(n + 64) & 127] * win;
    }
    kk[m] = v;
    sg[m] = noise_u[row * 512 + m] * 2.f - 1.f;
  }
  __syncthreads();
#pragma unroll
  for (int j = 0; j < 4; ++j) {                  // causal conv segment
    int i = tid + 128 * j;
    float acc = 0.f;
    for (int t = 0; t <= i; ++t) acc += kk[t] * sg[i - t];
    out[768000 + row * 512 + i] = acc;           // noise region of d_out
  }
}

// ------------------------------- reverb -------------------------------------
__global__ void k_imp(const float* __restrict__ rev_noise,
                      const float* __restrict__ decay,
                      const float* __restrict__ wet, float* __restrict__ imp) {
  int i = blockIdx.x * 256 + threadIdx.x;
  if (i >= 48128) return;
  float v = 0.f;
  if (i == 0) v = 1.f;
  else if (i < 48000) {
    float sp = log1pf(__expf(-decay[0]));        // softplus(-decay)
    float wf = 1.f / (1.f + __expf(-wet[0]));
    v = wf * rev_noise[i] * __expf(-sp * ((float)i / 48000.f) * 500.f);
  }
  imp[i] = v;
}

__global__ void k_addsig(const float* __restrict__ d_out, float* __restrict__ sig) {
  int t = blockIdx.x * 256 + threadIdx.x;
  if (t >= 768000) return;
  sig[t] = d_out[384000 + t] + d_out[768000 + t];
}

// out[b,i] = sum_{t<=min(i,47999)} imp[t] * sig[b, i-t]; LDS-staged chunks.
__global__ __launch_bounds__(256)
void k_reverb(const float* __restrict__ sig, const float* __restrict__ imp,
              float* __restrict__ out) {
  __shared__ float sw[512];
  __shared__ float si[256];
  int b  = blockIdx.y;
  int i0 = blockIdx.x * 256;
  int tid = threadIdx.x;
  const float* x = sig + (size_t)b * 192000;
  float acc = 0.f;
  int nch = i0 / 256 + 1;
  if (nch > 188) nch = 188;
  for (int c = 0; c < nch; ++c) {
    int tb = c * 256;
    int w0 = i0 - tb - 255;
    if (w0 >= 0) {                               // uniform branch across block
      if (tid < 128)       ddsp_copy16(x + w0 + tid * 4, &sw[tid * 4]);
      else if (tid < 192)  ddsp_copy16(imp + tb + (tid - 128) * 4, &si[(tid - 128) * 4]);
      ddsp_async_wait();
    } else {
      for (int j = tid; j < 512; j += 256) {
        int gx = w0 + j;
        sw[j] = (gx >= 0) ? x[gx] : 0.f;
      }
      si[tid] = imp[tb + tid];
    }
    __syncthreads();
    for (int dt = 0; dt < 256; ++dt)
      acc += si[dt] * sw[tid + 255 - dt];
    __syncthreads();
  }
  out[(size_t)b * 192000 + i0 + tid] = acc;
}

// ------------------------------ host orchestration --------------------------
extern "C" void kernel_launch(void* const* d_in, const int* in_sizes, int n_in,
                              void* d_out_v, int out_size, void* d_ws, size_t ws_size,
                              hipStream_t stream) {
  (void)in_sizes; (void)n_in; (void)out_size; (void)ws_size;
  auto in = [&](int i) { return (const float*)d_in[i]; };
  float* d_out = (float*)d_out_v;
  float* ws = (float*)d_ws;

  // workspace layout (floats)
  float* hidden1 = ws;                        // 750*1536
  float* tmpA    = ws + 1152000;              // 750*512
  float* tmpB    = ws + 1536000;              // 750*512
  float* GI      = ws + 1920000;              // 750*1536 (reused for p0/p1 raw)
  float* WhhT    = ws + 3072000;              // 512*1536
  float* H2      = ws + 3858432;              // 750*544
  float* ampsF   = ws + 4266432;              // 750*100
  float* P1s     = ws + 4341432;              // 750*65
  float* baseB   = ws + 4390182;              // 750
  float* impB    = ws + 4390932;              // 48128
  float* sigB    = ws + 4439060;              // 768000 (+slack)
  float* Wo1pad  = ws + 5207288;              // 512*544 zero-padded out_mlp.W1

  float* p0raw = GI;            // 750*101 (GI free after GRU)
  float* p1raw = GI + 200000;   // 750*65

  dim3 blk128(128), blk256(256);
  auto gemm = [&](const float* A, int lda, int Ka, const float* W, int Kw,
                  const float* bias, float* C, int ldc, int col0,
                  int M, int N, float c_add) {
    dim3 grid((M + 63) / 64, (N + 63) / 64);
    hipLaunchKernelGGL(k_gemm, grid, blk128, 0, stream,
                       A, lda, Ka, W, Kw, bias, C, ldc, col0, M, N, c_add);
  };
  auto ln = [&](const float* X, int ldx, const float* g, const float* be,
                float* Y, int ldy, int col0) {
    hipLaunchKernelGGL(k_ln, dim3(750), blk256, 0, stream, X, ldx, g, be, Y, ldy, col0);
  };

  const float* pitch = in(0);
  const float* loud  = in(1);

  // ---- weight prep (no dependencies)
  hipLaunchKernelGGL(k_padW, dim3((512*544+255)/256), blk256, 0, stream,
                     in(28), Wo1pad);
  hipLaunchKernelGGL(k_transp, dim3((1536*512+255)/256), blk256, 0, stream,
                     in(41), WhhT);

  // ---- mlp_pl(pitch) -> hidden1[:, 0:512]
  hipLaunchKernelGGL(k_affine1, dim3((750*512+255)/256), blk256, 0, stream,
                     pitch, in(4), in(5), tmpA);
  ln(tmpA, 512, in(6), in(7), tmpB, 512, 0);
  gemm(tmpB, 512, 512, in(8), 512, in(9), tmpA, 512, 0, 750, 512, 0.f);
  ln(tmpA, 512, in(10), in(11), tmpB, 512, 0);
  gemm(tmpB, 512, 512, in(12), 512, in(13), tmpA, 512, 0, 750, 512, 0.f);
  ln(tmpA, 512, in(14), in(15), hidden1, 1536, 0);

  // ---- mlp_pl(loudness) -> hidden1[:, 512:1024]
  hipLaunchKernelGGL(k_affine1, dim3((750*512+255)/256), blk256, 0, stream,
                     loud, in(4), in(5), tmpA);
  ln(tmpA, 512, in(6), in(7), tmpB, 512, 0);
  gemm(tmpB, 512, 512, in(8), 512, in(9), tmpA, 512, 0, 750, 512, 0.f);
  ln(tmpA, 512, in(10), in(11), tmpB, 512, 0);
  gemm(tmpB, 512, 512, in(12), 512, in(13), tmpA, 512, 0, 750, 512, 0.f);
  ln(tmpA, 512, in(14), in(15), hidden1, 1536, 512);

  // ---- mlp_z(z) -> hidden1[:, 1024:1536]
  hipLaunchKernelGGL(k_gemm16, dim3((750*512+255)/256), blk256, 0, stream,
                     in(2), in(16), in(17), tmpA);
  ln(tmpA, 512, in(18), in(19), tmpB, 512, 0);
  gemm(tmpB, 512, 512, in(20), 512, in(21), tmpA, 512, 0, 750, 512, 0.f);
  ln(tmpA, 512, in(22), in(23), tmpB, 512, 0);
  gemm(tmpB, 512, 512, in(24), 512, in(25), tmpA, 512, 0, 750, 512, 0.f);
  ln(tmpA, 512, in(26), in(27), hidden1, 1536, 1024);

  // ---- GRU: GI = hidden1 @ W_ih^T + b_ih (WMMA), then serial recurrence
  gemm(hidden1, 1536, 1536, in(40), 1536, in(42), GI, 1536, 0, 750, 1536, 0.f);
  hipLaunchKernelGGL(k_gru, dim3(2), dim3(768), 0, stream,
                     GI, WhhT, in(43), pitch, loud, H2);

  // ---- out_mlp (K padded 514 -> 544 on both operands)
  gemm(H2, 544, 544, Wo1pad, 544, in(29), tmpA, 512, 0, 750, 512, 0.f);
  ln(tmpA, 512, in(30), in(31), tmpB, 512, 0);
  gemm(tmpB, 512, 512, in(32), 512, in(33), tmpA, 512, 0, 750, 512, 0.f);
  ln(tmpA, 512, in(34), in(35), tmpB, 512, 0);
  gemm(tmpB, 512, 512, in(36), 512, in(37), tmpA, 512, 0, 750, 512, 0.f);
  ln(tmpA, 512, in(38), in(39), tmpB, 512, 0);       // final hidden in tmpB

  // ---- projections
  gemm(tmpB, 512, 512, in(44), 512, in(45), p0raw, 101, 0, 750, 101, 0.f);
  gemm(tmpB, 512, 512, in(46), 512, in(47), p1raw, 65, 0, 750, 65, -5.f);
  hipLaunchKernelGGL(k_proj0, dim3(750), blk128, 0, stream,
                     p0raw, pitch, ampsF, d_out + 1152000);
  hipLaunchKernelGGL(k_proj1, dim3((750*65+255)/256), blk256, 0, stream,
                     p1raw, P1s);

  // ---- synthesis
  hipLaunchKernelGGL(k_base, dim3(1), dim3(32), 0, stream, pitch, baseB);
  hipLaunchKernelGGL(k_harm, dim3(750), blk256, 0, stream,
                     pitch, baseB, ampsF, d_out);
  hipLaunchKernelGGL(k_noise, dim3(750), blk128, 0, stream,
                     in(3), P1s, d_out);

  // ---- reverb
  hipLaunchKernelGGL(k_imp, dim3((48128+255)/256), blk256, 0, stream,
                     in(48), in(49), in(50), impB);
  hipLaunchKernelGGL(k_addsig, dim3((768000+255)/256), blk256, 0, stream,
                     d_out, sigB);
  hipLaunchKernelGGL(k_reverb, dim3(750, 2), blk256, 0, stream,
                     sigB, impB, d_out);
}
